// MultiModalSDTPS_28080496181363
// MI455X (gfx1250) — compile-verified, matrix operand order checked by reading, not run on hardware
//
#include <hip/hip_runtime.h>
#include <hip/hip_bf16.h>
#include <math.h>

typedef __attribute__((ext_vector_type(16))) _Float16 v16h;
typedef __attribute__((ext_vector_type(8)))  float    v8f;
typedef __attribute__((ext_vector_type(4)))  int      v4i;

#define EMBED   512
#define NTOK    256
#define NBATCH  64
#define COS_TAU 0.3f
#define SOFT_TAU 0.3f
#define SCALEQK 0.044194173824159216f  /* 512^-0.5 */

// gfx1250 async global->LDS copy path (ASYNCcnt-tracked), guarded so the file
// still compiles if the toolchain lacks the builtins.
#if defined(__AMDGCN__) &&                                            \
    __has_builtin(__builtin_amdgcn_global_load_async_to_lds_b128) &&  \
    __has_builtin(__builtin_amdgcn_s_wait_asynccnt)
#define HAVE_ASYNC_LDS 1
// builtin expects (v4i AS1* src, v4i AS3* dst, imm offset, imm cpol)
#define ASYNC_GP(x) ((__attribute__((address_space(1))) v4i*)(x))
#define ASYNC_LP(x) ((__attribute__((address_space(3))) v4i*)(x))
#endif

// guide global index used by each of the 9 attention modules
__device__ __constant__ int c_gsel[9] = {0, 1, 2, 1, 0, 2, 2, 0, 1};
// per-modality guide order (self, other1, other2)
__device__ __constant__ int c_guide[3][3] = {{0, 1, 2}, {1, 0, 2}, {2, 0, 1}};

__device__ inline float dot4(float4 a, float4 b) {
  return a.x * b.x + a.y * b.y + a.z * b.z + a.w * b.w;
}

__device__ inline float gelu_exact(float x) {
  return 0.5f * x * (1.0f + erff(x * 0.70710678118654752f));
}

__device__ inline float block_sum256(float v, float* red) {
  const int t = threadIdx.x;
  red[t] = v;
  __syncthreads();
  for (int s = 128; s > 0; s >>= 1) {
    if (t < s) red[t] += red[t + s];
    __syncthreads();
  }
  const float r = red[0];
  __syncthreads();
  return r;
}

__device__ inline float block_max256(float v, float* red) {
  const int t = threadIdx.x;
  red[t] = v;
  __syncthreads();
  for (int s = 128; s > 0; s >>= 1) {
    if (t < s) red[t] = fmaxf(red[t], red[t + s]);
    __syncthreads();
  }
  const float r = red[0];
  __syncthreads();
  return r;
}

// ---------------------------------------------------------------------------
// Kernel 1: per attention module j (9 blocks x 16 waves):
//   Q = G_guide @ QW_j + qb_j   (64x512 @ 512x512) -> LDS (f16)
//   qkb_j[b] = Q[b,:] . kb_j
//   V_j = Q @ KW_j^T            (64x512 @ 512x512) -> global ws
// v_wmma_f32_16x16x32_f16 with the B k-slab staged into LDS per k-step via
// async global->LDS (ASYNCcnt), all 16 waves consuming the shared slab.
// Each wave owns 8 N-tiles of one M-tile (acc in 64 VGPRs), A frag loaded once
// per k-step. Fragment layouts per CDNA5 ISA (wave32).
// ---------------------------------------------------------------------------
__global__ __launch_bounds__(512) void qv_wmma_kernel(
    const float* __restrict__ g0, const float* __restrict__ g1,
    const float* __restrict__ g2, const float* __restrict__ qw,
    const float* __restrict__ qb, const float* __restrict__ kw,
    const float* __restrict__ kb, float* __restrict__ vbuf,
    float* __restrict__ qkbbuf) {
  __shared__ _Float16 sQ[NBATCH * EMBED];  // 64 KB
  __shared__ float sB[32 * EMBED];         // 64 KB k-slab (shared B operand)

  const int tid  = threadIdx.x;
  const int lane = tid & 31;
  const int wave = tid >> 5;
  const int j    = blockIdx.x;
  const int gs   = c_gsel[j];
  const float* G  = (gs == 0) ? g0 : (gs == 1) ? g1 : g2;
  const float* QW = qw + (size_t)j * EMBED * EMBED;
  const float* KW = kw + (size_t)j * EMBED * EMBED;
  const int ln  = lane & 15;
  const int hi8 = (lane & 16) ? 8 : 0;  // +8 K (A/B frags) / +8 M (C/D frags)
  const int mt  = wave >> 2;            // M-tile of this wave (0..3)
  const int nt0 = (wave & 3) * 8;       // first of 8 N-tiles of this wave
  const int r   = mt * 16 + ln;         // A-frag row

  v8f acc[8];

  // ==================== GEMM 1: Q = G @ QW + qb ====================
#pragma unroll
  for (int t8 = 0; t8 < 8; ++t8) {
    const float bias = qb[j * EMBED + (nt0 + t8) * 16 + ln];
#pragma unroll
    for (int v = 0; v < 8; ++v) acc[t8][v] = bias;
  }

  for (int kk = 0; kk < EMBED; kk += 32) {
    // stage QW rows [kk, kk+32) -> sB[ks*512 + c], fully coalesced b128s
#if HAVE_ASYNC_LDS
    for (int i = tid; i < 32 * EMBED / 4; i += 512)
      __builtin_amdgcn_global_load_async_to_lds_b128(
          ASYNC_GP(QW + (size_t)kk * EMBED + i * 4), ASYNC_LP(sB + i * 4), 0, 0);
    __builtin_amdgcn_s_wait_asynccnt(0);
#else
    for (int i = tid; i < 32 * EMBED / 4; i += 512)
      ((float4*)sB)[i] = ((const float4*)(QW + (size_t)kk * EMBED))[i];
#endif
    __syncthreads();

    v16h a;
#pragma unroll
    for (int v = 0; v < 8; ++v) {
      const int k = kk + hi8 + ((v & 4) ? 16 : 0) + ((v & 3) << 1);
      a[2 * v]     = (_Float16)G[r * EMBED + k];
      a[2 * v + 1] = (_Float16)G[r * EMBED + k + 1];
    }
#pragma unroll
    for (int t8 = 0; t8 < 8; ++t8) {
      const int n = (nt0 + t8) * 16 + ln;
      v16h bf;
#pragma unroll
      for (int v = 0; v < 8; ++v) {
        const int ks = hi8 + ((v & 4) ? 16 : 0) + ((v & 3) << 1);
        bf[2 * v]     = (_Float16)sB[ks * EMBED + n];
        bf[2 * v + 1] = (_Float16)sB[(ks + 1) * EMBED + n];
      }
      acc[t8] = __builtin_amdgcn_wmma_f32_16x16x32_f16(false, a, false, bf,
                                                       (short)0, acc[t8],
                                                       false, false);
    }
    __syncthreads();  // slab consumed; safe to refill next k-step
  }

#pragma unroll
  for (int t8 = 0; t8 < 8; ++t8) {
    const int n = (nt0 + t8) * 16 + ln;
#pragma unroll
    for (int v = 0; v < 8; ++v)
      sQ[(mt * 16 + hi8 + v) * EMBED + n] = (_Float16)acc[t8][v];
  }
  __syncthreads();

  // ==================== qkb[b] = Q[b,:] . kb ====================
  if (tid < NBATCH) {
    float s = 0.f;
    for (int c = 0; c < EMBED; ++c)
      s += (float)sQ[tid * EMBED + c] * kb[j * EMBED + c];
    qkbbuf[j * NBATCH + tid] = s;
  }
  __syncthreads();

  // ==================== GEMM 2: V = Q @ KW^T ====================
  // slab layout [n][ks]: sB[n*32 + ks] = KW[n*512 + kk + ks]
#pragma unroll
  for (int t8 = 0; t8 < 8; ++t8)
#pragma unroll
    for (int v = 0; v < 8; ++v) acc[t8][v] = 0.f;

  for (int kk = 0; kk < EMBED; kk += 32) {
#if HAVE_ASYNC_LDS
    for (int i = tid; i < 512 * 8; i += 512) {
      const int n = i >> 3, q4 = i & 7;
      __builtin_amdgcn_global_load_async_to_lds_b128(
          ASYNC_GP(KW + (size_t)n * EMBED + kk + q4 * 4),
          ASYNC_LP(sB + n * 32 + q4 * 4), 0, 0);
    }
    __builtin_amdgcn_s_wait_asynccnt(0);
#else
    for (int i = tid; i < 512 * 8; i += 512) {
      const int n = i >> 3, q4 = i & 7;
      *(float4*)(sB + n * 32 + q4 * 4) =
          *(const float4*)(KW + (size_t)n * EMBED + kk + q4 * 4);
    }
#endif
    __syncthreads();

    v16h a;
#pragma unroll
    for (int v = 0; v < 8; ++v) {
      const int k = kk + hi8 + ((v & 4) ? 16 : 0) + ((v & 3) << 1);
      a[2 * v]     = sQ[r * EMBED + k];
      a[2 * v + 1] = sQ[r * EMBED + k + 1];
    }
#pragma unroll
    for (int t8 = 0; t8 < 8; ++t8) {
      const int n = (nt0 + t8) * 16 + ln;
      v16h bf;
#pragma unroll
      for (int v = 0; v < 8; ++v) {
        const int ks = hi8 + ((v & 4) ? 16 : 0) + ((v & 3) << 1);
        bf[2 * v]     = (_Float16)sB[n * 32 + ks];
        bf[2 * v + 1] = (_Float16)sB[n * 32 + ks + 1];
      }
      acc[t8] = __builtin_amdgcn_wmma_f32_16x16x32_f16(false, a, false, bf,
                                                       (short)0, acc[t8],
                                                       false, false);
    }
    __syncthreads();
  }

#pragma unroll
  for (int t8 = 0; t8 < 8; ++t8) {
    const int n = (nt0 + t8) * 16 + ln;
#pragma unroll
    for (int v = 0; v < 8; ++v)
      vbuf[((size_t)j * NBATCH + mt * 16 + hi8 + v) * EMBED + n] = acc[t8][v];
  }
}

// ---------------------------------------------------------------------------
// Kernel 2: adaptive-weight MLP per modality (3 blocks x 256 threads).
// ---------------------------------------------------------------------------
__global__ __launch_bounds__(256) void mlp_weights_kernel(
    const float* __restrict__ g0, const float* __restrict__ g1,
    const float* __restrict__ g2, const float* __restrict__ w1,
    const float* __restrict__ b1, const float* __restrict__ lng,
    const float* __restrict__ lnb, const float* __restrict__ w2,
    const float* __restrict__ b2, const float* __restrict__ w3,
    const float* __restrict__ b3, float* __restrict__ mlpw) {
  __shared__ float red[256];
  __shared__ float h1[256];
  __shared__ float h2[64];
  const int mi = blockIdx.x, t = threadIdx.x;
  const float* W1 = w1 + (size_t)mi * 1536 * 256;
  const float* W2 = w2 + (size_t)mi * 256 * 64;
  const float* W3 = w3 + (size_t)mi * 64 * 3;

  for (int b = 0; b < NBATCH; ++b) {
    const float* ga = g0 + b * EMBED;
    const float* gb = g1 + b * EMBED;
    const float* gc = g2 + b * EMBED;
    float acc = b1[mi * 256 + t];
    for (int k = 0; k < EMBED; ++k) acc += ga[k] * W1[(size_t)k * 256 + t];
    for (int k = 0; k < EMBED; ++k) acc += gb[k] * W1[(size_t)(512 + k) * 256 + t];
    for (int k = 0; k < EMBED; ++k) acc += gc[k] * W1[(size_t)(1024 + k) * 256 + t];
    const float mean = block_sum256(acc, red) * (1.f / 256.f);
    const float var  = block_sum256(acc * acc, red) * (1.f / 256.f) - mean * mean;
    float y = (acc - mean) / sqrtf(var + 1e-5f) * lng[mi * 256 + t] + lnb[mi * 256 + t];
    h1[t] = gelu_exact(y);
    __syncthreads();
    if (t < 64) {
      float a2 = b2[mi * 64 + t];
      for (int k = 0; k < 256; ++k) a2 += h1[k] * W2[k * 64 + t];
      h2[t] = gelu_exact(a2);
    }
    __syncthreads();
    if (t == 0) {
      float o[3];
      float mx = -1e30f;
      for (int oi = 0; oi < 3; ++oi) {
        float a3 = b3[mi * 3 + oi];
        for (int k = 0; k < 64; ++k) a3 += h2[k] * W3[k * 3 + oi];
        o[oi] = a3;
        mx = fmaxf(mx, a3);
      }
      float se = 0.f;
      for (int oi = 0; oi < 3; ++oi) { o[oi] = expf(o[oi] - mx); se += o[oi]; }
      for (int oi = 0; oi < 3; ++oi)
        mlpw[(mi * NBATCH + b) * 3 + oi] = o[oi] / se;
    }
    __syncthreads();
  }
}

// ---------------------------------------------------------------------------
// Kernel 3: fused scoring per (modality, batch) — 192 blocks x 256 threads.
// ---------------------------------------------------------------------------
__global__ __launch_bounds__(256) void score_mask_kernel(
    const float* __restrict__ rgb, const float* __restrict__ nir,
    const float* __restrict__ tir, const float* __restrict__ g0,
    const float* __restrict__ g1, const float* __restrict__ g2,
    const float* __restrict__ vbuf, const float* __restrict__ qkbbuf,
    const float* __restrict__ mlpw, float* __restrict__ out) {
  __shared__ float red[256];
  __shared__ float score[256];
  __shared__ float maskbuf[256];
  __shared__ float gnorm[3];
  __shared__ float thr;

  const int t  = threadIdx.x;
  const int mi = blockIdx.x >> 6;
  const int b  = blockIdx.x & 63;
  const float* pats = (mi == 0) ? rgb : (mi == 1) ? nir : tir;
  const float* gls[3] = {g0, g1, g2};

  if (t < 3) {
    const float* g = gls[t] + b * EMBED;
    float s = 0.f;
    for (int c = 0; c < EMBED; ++c) s += g[c] * g[c];
    gnorm[t] = fmaxf(sqrtf(s), 1e-8f);
  }
  __syncthreads();

  const float* prow = pats + ((size_t)b * NTOK + t) * EMBED;
  __builtin_prefetch(prow, 0, 1);  // global_prefetch_b8
  const int gA = c_guide[mi][0], gB = c_guide[mi][1], gC = c_guide[mi][2];
  const float4* p4 = (const float4*)prow;
  const float4* v0 = (const float4*)(vbuf + ((size_t)(3 * mi + 0) * NBATCH + b) * EMBED);
  const float4* v1 = (const float4*)(vbuf + ((size_t)(3 * mi + 1) * NBATCH + b) * EMBED);
  const float4* v2 = (const float4*)(vbuf + ((size_t)(3 * mi + 2) * NBATCH + b) * EMBED);
  const float4* ha = (const float4*)(gls[gA] + b * EMBED);
  const float4* hb = (const float4*)(gls[gB] + b * EMBED);
  const float4* hc = (const float4*)(gls[gC] + b * EMBED);

  float dv0 = 0.f, dv1 = 0.f, dv2 = 0.f, da = 0.f, db = 0.f, dc = 0.f, pn2 = 0.f;
  for (int c = 0; c < EMBED / 4; ++c) {
    const float4 p = p4[c];
    dv0 += dot4(p, v0[c]);
    dv1 += dot4(p, v1[c]);
    dv2 += dot4(p, v2[c]);
    da  += dot4(p, ha[c]);
    db  += dot4(p, hb[c]);
    dc  += dot4(p, hc[c]);
    pn2 += dot4(p, p);
  }
  const float pn = fmaxf(sqrtf(pn2), 1e-8f);

  float lg[3];
  lg[0] = (dv0 + qkbbuf[(3 * mi + 0) * NBATCH + b]) * SCALEQK + (da / (pn * gnorm[gA])) * (1.f / COS_TAU);
  lg[1] = (dv1 + qkbbuf[(3 * mi + 1) * NBATCH + b]) * SCALEQK + (db / (pn * gnorm[gB])) * (1.f / COS_TAU);
  lg[2] = (dv2 + qkbbuf[(3 * mi + 2) * NBATCH + b]) * SCALEQK + (dc / (pn * gnorm[gC])) * (1.f / COS_TAU);

  float snr[3];
  for (int jj = 0; jj < 3; ++jj) {
    const float mx = block_max256(lg[jj], red);
    const float e  = expf(lg[jj] - mx);
    const float s  = e / block_sum256(e, red);
    const float mean = block_sum256(s, red) * (1.f / 256.f);
    const float sq   = block_sum256(s * s, red);
    const float var  = (sq - 256.f * mean * mean) * (1.f / 255.f);
    const float sd   = sqrtf(fmaxf(var, 0.f)) + 1e-5f;
    snr[jj] = 1.f / (1.f + expf(-(s - mean) / sd));
  }

  const float w0 = mlpw[(mi * NBATCH + b) * 3 + 0];
  const float w1 = mlpw[(mi * NBATCH + b) * 3 + 1];
  const float w2 = mlpw[(mi * NBATCH + b) * 3 + 2];
  const float sc = w0 * snr[0] + w1 * snr[1] + w2 * snr[2];
  score[t] = sc;
  __syncthreads();

  // exact rank selection: thr = sorted(score)[102]  (0.4 * 255 = 102)
  int rnk = 0;
  for (int m2 = 0; m2 < 256; ++m2) {
    const float o = score[m2];
    rnk += (o < sc) || (o == sc && m2 < t);
  }
  if (rnk == 102) thr = sc;
  __syncthreads();

  const float mask = 1.f / (1.f + expf(-(sc - thr) * (1.f / SOFT_TAU)));
  maskbuf[t] = mask;
  const size_t TOKOUT = (size_t)3 * NBATCH * NTOK * EMBED;
  out[TOKOUT + ((size_t)mi * NBATCH + b) * NTOK + t] = mask;
  __syncthreads();

  float* op = out + ((size_t)mi * NBATCH + b) * NTOK * EMBED;
  const float* pb = pats + (size_t)b * NTOK * EMBED;
  for (int idx = t; idx < NTOK * EMBED; idx += 256)
    op[idx] = pb[idx] * maskbuf[idx >> 9];
}

extern "C" void kernel_launch(void* const* d_in, const int* in_sizes, int n_in,
                              void* d_out, int out_size, void* d_ws,
                              size_t ws_size, hipStream_t stream) {
  (void)in_sizes; (void)n_in; (void)out_size; (void)ws_size;
  const float* rgb  = (const float*)d_in[0];
  const float* nir  = (const float*)d_in[1];
  const float* tir  = (const float*)d_in[2];
  const float* rgbg = (const float*)d_in[3];
  const float* nirg = (const float*)d_in[4];
  const float* tirg = (const float*)d_in[5];
  const float* aqw  = (const float*)d_in[6];
  const float* aqb  = (const float*)d_in[7];
  const float* akw  = (const float*)d_in[8];
  const float* akb  = (const float*)d_in[9];
  const float* w1   = (const float*)d_in[10];
  const float* b1   = (const float*)d_in[11];
  const float* lng  = (const float*)d_in[12];
  const float* lnb  = (const float*)d_in[13];
  const float* w2   = (const float*)d_in[14];
  const float* b2   = (const float*)d_in[15];
  const float* w3   = (const float*)d_in[16];
  const float* b3   = (const float*)d_in[17];
  float* out = (float*)d_out;

  float* ws     = (float*)d_ws;
  float* vbuf   = ws;                         // 9*64*512 f32
  float* qkbbuf = vbuf + 9 * NBATCH * EMBED;  // 9*64 f32
  float* mlpw   = qkbbuf + 9 * NBATCH;        // 3*64*3 f32

  qv_wmma_kernel<<<9, 512, 0, stream>>>(rgbg, nirg, tirg, aqw, aqb, akw, akb,
                                        vbuf, qkbbuf);
  mlp_weights_kernel<<<3, 256, 0, stream>>>(rgbg, nirg, tirg, w1, b1, lng, lnb,
                                            w2, b2, w3, b3, mlpw);
  score_mask_kernel<<<192, 256, 0, stream>>>(rgb, nir, tir, rgbg, nirg, tirg,
                                             vbuf, qkbbuf, mlpw, out);
}
